// VectorQuantizerEMA_5875515261472
// MI455X (gfx1250) — matrix operand order repaired and checked by reading, not compiled, not run on hardware
//
#include <hip/hip_runtime.h>

// ---------------------------------------------------------------------------
// VectorQuantizerEMA for MI455X (gfx1250, wave32, WMMA)
// N=65536 rows, D=256 features, K=4096 codes
// ---------------------------------------------------------------------------

#define N_ROWS 65536
#define DFEAT  256
#define KCODES 4096
#define EPS    1e-5f
#define DECAY  0.99f
#define OMD    0.01f      // 1 - DECAY
#define BNM    0.1f       // BN momentum

typedef __bf16 v16bf __attribute__((ext_vector_type(16)));
typedef float  v8f   __attribute__((ext_vector_type(8)));
typedef unsigned int v8u __attribute__((ext_vector_type(8)));

// ---- workspace layout (floats unless noted) -------------------------------
#define WS_SUM    0            // [D]  BN column sums
#define WS_SSQ    256          // [D]  BN column sum-of-squares
#define WS_MEAN   512          // [D]
#define WS_ISTD   768          // [D]  1/sqrt(var+eps)
#define WS_RSTD   1024         // [D]  sqrt(new_running_var+eps)
#define WS_NTOT   1280         // [1]  scalar n_total
#define WS_CNT    1536         // [K]  counts
#define WS_CSPRE  (1536+4096)  // [K]  cs * decay + (1-decay)*counts
#define WS_ENORM  (1536+8192)  // [K]  ||embedding_k||^2
#define WS_HEADF  (1536+12288) // head size in floats (13824)
#define OFF_XN    65536        // byte offset: swizzled bf16 xn, N*D*2 = 32MB
#define OFF_EB    (65536 + 33554432) // byte offset: swizzled bf16 embedding, 2MB

// ---- d_out layout (floats, concatenated in return order) ------------------
#define OUT_IDX    0
#define OUT_EMB    (N_ROWS)                       // new_embedding  [K,D]
#define OUT_EMBO   (N_ROWS + 1048576)             // embedding_output [K,D]
#define OUT_CS     (N_ROWS + 2097152)             // new_cs [K]
#define OUT_EMAW   (N_ROWS + 2097152 + KCODES)    // new_ema_w [K,D]
#define OUT_RM     (N_ROWS + 3145728 + KCODES)    // new_running_mean [D]
#define OUT_RV     (N_ROWS + 3145728 + KCODES + DFEAT) // new_running_var [D]

__device__ __forceinline__ void atomAddF(float* p, float v) {
  __hip_atomic_fetch_add(p, v, __ATOMIC_RELAXED, __HIP_MEMORY_SCOPE_AGENT);
}

// round-to-nearest-even pack of two f32 into packed bf16x2
__device__ __forceinline__ unsigned bf16pack(float a, float b) {
  unsigned ua = __builtin_bit_cast(unsigned, a);
  unsigned ub = __builtin_bit_cast(unsigned, b);
  ua = (ua + 0x7FFFu + ((ua >> 16) & 1u)) >> 16;
  ub = (ub + 0x7FFFu + ((ub >> 16) & 1u));
  return (ua & 0xFFFFu) | (ub & 0xFFFF0000u);
}
__device__ __forceinline__ float bf16lo(unsigned u) {
  return __builtin_bit_cast(float, u << 16);
}
__device__ __forceinline__ float bf16hi(unsigned u) {
  return __builtin_bit_cast(float, u & 0xFFFF0000u);
}

// ---------------------------------------------------------------------------
// 1) zero workspace head
__global__ void k_init_ws(float* ws) {
  int i = blockIdx.x * 256 + threadIdx.x;
  if (i < WS_HEADF) ws[i] = 0.0f;
}

// 2) BN column stats: each block reduces 256 rows, coalesced, then atomics
__global__ void k_bn_stats(const float* __restrict__ x, float* ws) {
  int d  = threadIdx.x;
  int r0 = blockIdx.x * 256;
  float s = 0.f, ss = 0.f;
  for (int r = 0; r < 256; ++r) {
    float v = x[(size_t)(r0 + r) * DFEAT + d];
    s += v; ss += v * v;
  }
  atomAddF(&ws[WS_SUM + d], s);
  atomAddF(&ws[WS_SSQ + d], ss);
}

// 3) finalize BN stats + running stats outputs
__global__ void k_bn_finalize(float* ws, const float* __restrict__ rmean,
                              const float* __restrict__ rvar, float* out) {
  int d = threadIdx.x;
  const float invN = 1.0f / (float)N_ROWS;
  float mean = ws[WS_SUM + d] * invN;
  float var  = ws[WS_SSQ + d] * invN - mean * mean;
  ws[WS_MEAN + d] = mean;
  ws[WS_ISTD + d] = __frsqrt_rn(var + EPS);
  float nrm = (1.0f - BNM) * rmean[d] + BNM * mean;
  float varu = var * ((float)N_ROWS / (float)(N_ROWS - 1));
  float nrv = (1.0f - BNM) * rvar[d] + BNM * varu;
  out[OUT_RM + d] = nrm;
  out[OUT_RV + d] = nrv;
  ws[WS_RSTD + d] = sqrtf(nrv + EPS);
}

// 4) normalize x and store bf16 as WMMA A fragments in this code's contiguous k order
//    A 16x32 bf16 tile: lane = m + ((kk>>4)&1)*16 ; elem = kk&15
__global__ void k_swizzle_x(const float* __restrict__ x, const float* ws,
                            unsigned char* wsb) {
  int id = blockIdx.x * 256 + threadIdx.x;    // N * D/8 = 2M threads
  int n = id >> 5, j = id & 31;               // j: 8-element group along D
  const float4* x4 = (const float4*)(x + (size_t)n * DFEAT + j * 8);
  float4 p = x4[0], q = x4[1];
  float f[8] = {p.x, p.y, p.z, p.w, q.x, q.y, q.z, q.w};
  int d0 = j * 8;
  unsigned o[4];
#pragma unroll
  for (int i = 0; i < 4; ++i) {
    float a = (f[2 * i]     - ws[WS_MEAN + d0 + 2 * i])     * ws[WS_ISTD + d0 + 2 * i];
    float b = (f[2 * i + 1] - ws[WS_MEAN + d0 + 2 * i + 1]) * ws[WS_ISTD + d0 + 2 * i + 1];
    o[i] = bf16pack(a, b);
  }
  int t = n >> 4, m = n & 15, c = j >> 2, g = j & 3;
  int lane = m + (g >> 1) * 16;
  int e0 = (g & 1) ? 8 : 0;
  size_t elem = ((size_t)(t * 8 + c) * 32 + lane) * 16 + e0;
  uint4 u4; u4.x = o[0]; u4.y = o[1]; u4.z = o[2]; u4.w = o[3];
  *(uint4*)(wsb + OFF_XN + elem * 2) = u4;
}

// 5) swizzle embedding into B fragments (same contiguous k order as A) + accumulate ||e_k||^2
//    B 32x16 bf16 tile: lane = (k&15) + hi*16 ; elems = 16 contiguous d
__global__ void k_swizzle_e(const float* __restrict__ e, float* ws,
                            unsigned char* wsb) {
  int id = blockIdx.x * 256 + threadIdx.x;    // K * D/16 = 64K threads
  int k = id >> 4, j2 = id & 15;              // j2: 16-element group along D
  const float4* e4 = (const float4*)(e + (size_t)k * DFEAT + j2 * 16);
  float4 v0 = e4[0], v1 = e4[1], v2 = e4[2], v3 = e4[3];
  float ss = v0.x*v0.x + v0.y*v0.y + v0.z*v0.z + v0.w*v0.w
           + v1.x*v1.x + v1.y*v1.y + v1.z*v1.z + v1.w*v1.w
           + v2.x*v2.x + v2.y*v2.y + v2.z*v2.z + v2.w*v2.w
           + v3.x*v3.x + v3.y*v3.y + v3.z*v3.z + v3.w*v3.w;
  atomAddF(&ws[WS_ENORM + k], ss);
  unsigned o[8] = {
    bf16pack(v0.x, v0.y), bf16pack(v0.z, v0.w),
    bf16pack(v1.x, v1.y), bf16pack(v1.z, v1.w),
    bf16pack(v2.x, v2.y), bf16pack(v2.z, v2.w),
    bf16pack(v3.x, v3.y), bf16pack(v3.z, v3.w)};
  int u = k >> 4, ll = k & 15, c = j2 >> 1, hi = j2 & 1;
  int lane = ll + hi * 16;
  size_t elem = ((size_t)(u * 8 + c) * 32 + lane) * 16;
  uint4 a; a.x = o[0]; a.y = o[1]; a.z = o[2]; a.w = o[3];
  uint4 b; b.x = o[4]; b.y = o[5]; b.z = o[6]; b.w = o[7];
  *(uint4*)(wsb + OFF_EB + elem * 2) = a;
  *(uint4*)(wsb + OFF_EB + elem * 2 + 16) = b;
}

// 6) initialize new_ema_w region to DECAY * ema_w (scatter target)
__global__ void k_ema_init(const float* __restrict__ emaw, float* out) {
  int id = blockIdx.x * 256 + threadIdx.x;    // K*D = 1M threads
  out[OUT_EMAW + id] = DECAY * emaw[id];
}

// 7) WMMA GEMM + fused argmin with LDS-staged B (shared by all 8 waves).
//    One wave = 16 rows x all K. argmin(rn + cn - 2*dot) == argmin(cn - 2*dot).
//    Stage = 4 column tiles = 32KB of B; double-buffered (64KB LDS/block).
//    Pipeline: global->regs for stage s+1 overlaps LDS compute of stage s.
#define STAGE_TILES 4
#define STAGE_U4    2048          /* 32KB per stage in uint4 units */
__global__ __launch_bounds__(256) void k_gemm_argmin(
    const unsigned char* __restrict__ wsb, const float* __restrict__ ws,
    float* __restrict__ out) {
  const int lane = threadIdx.x;               // 0..31
  const int wave = threadIdx.y;               // 0..7
  const int thr  = wave * 32 + lane;          // 0..255
  const int t = blockIdx.x * 8 + wave;        // row tile (16 rows)

  __shared__ __align__(16) uint4 sbuf[2][STAGE_U4];

  const v8u* __restrict__ abase = (const v8u*)(wsb + OFF_XN);
  const uint4* __restrict__ gB = (const uint4*)(wsb + OFF_EB);
  const float* __restrict__ enorm = ws + WS_ENORM;

  // A fragments for this wave's 16 rows: full D=256, register resident
  v8u a[8];
#pragma unroll
  for (int c = 0; c < 8; ++c) a[c] = abase[(t * 8 + c) * 32 + lane];

  float minv[8]; int mini[8];
#pragma unroll
  for (int r = 0; r < 8; ++r) { minv[r] = 3.4e38f; mini[r] = 0; }

  const int nsub = lane & 15;
  const int NSTAGE = (KCODES / 16) / STAGE_TILES;   // 64

  // prologue: stage 0 into buffer 0
  uint4 pre[8];
#pragma unroll
  for (int k = 0; k < 8; ++k) pre[k] = gB[thr + 256 * k];
#pragma unroll
  for (int k = 0; k < 8; ++k) sbuf[0][thr + 256 * k] = pre[k];
  __syncthreads();

  int p = 0;
  for (int s = 0; s < NSTAGE; ++s) {
    // issue next stage's global loads (results consumed after compute)
    if (s + 1 < NSTAGE) {
      const uint4* g = gB + (size_t)(s + 1) * STAGE_U4;
#pragma unroll
      for (int k = 0; k < 8; ++k) pre[k] = g[thr + 256 * k];
    }
    // compute 4 column tiles from LDS buffer p
    const uint4* sb = &sbuf[p][0];
#pragma unroll
    for (int uu = 0; uu < STAGE_TILES; ++uu) {
      int u = s * STAGE_TILES + uu;
      float cn = enorm[u * 16 + nsub];
      v8f acc = {};
#pragma unroll
      for (int f = 0; f < 8; ++f) {
        int b4 = ((uu * 8 + f) * 32 + lane) * 2;    // uint4 index
        uint4 lo = sb[b4], hi = sb[b4 + 1];
        v8u bu = {lo.x, lo.y, lo.z, lo.w, hi.x, hi.y, hi.z, hi.w};
        v16bf aa = __builtin_bit_cast(v16bf, a[f]);
        v16bf bb = __builtin_bit_cast(v16bf, bu);
        acc = __builtin_amdgcn_wmma_f32_16x16x32_bf16(
            false, aa, false, bb, (short)0, acc, false, false);
      }
      int col = u * 16 + nsub;
#pragma unroll
      for (int r = 0; r < 8; ++r) {
        float dst = cn - 2.0f * acc[r];
        bool lt = dst < minv[r];
        minv[r] = lt ? dst : minv[r];
        mini[r] = lt ? col : mini[r];
      }
    }
    __syncthreads();   // all waves done reading buffer (1-p) two stages back
    if (s + 1 < NSTAGE) {
#pragma unroll
      for (int k = 0; k < 8; ++k) sbuf[1 - p][thr + 256 * k] = pre[k];
    }
    __syncthreads();   // stage s+1 visible to all waves
    p ^= 1;
  }

  // reduce (min,argmin) across 16 lanes sharing the same rows
#pragma unroll
  for (int m = 8; m >= 1; m >>= 1) {
#pragma unroll
    for (int r = 0; r < 8; ++r) {
      float ov = __shfl_xor(minv[r], m, 32);
      int   oi = __shfl_xor(mini[r], m, 32);
      if (ov < minv[r] || (ov == minv[r] && oi < mini[r])) {
        minv[r] = ov; mini[r] = oi;
      }
    }
  }
  if (nsub == 0) {
    int rowbase = t * 16 + ((lane >= 16) ? 8 : 0);
#pragma unroll
    for (int r = 0; r < 8; ++r)
      out[OUT_IDX + rowbase + r] = (float)mini[r];
  }
}

// 8) scatter: counts and (1-decay)*xn into new_ema_w region
__global__ void k_scatter(const unsigned char* __restrict__ wsb, float* ws,
                          float* out) {
  int id = blockIdx.x * 256 + threadIdx.x;    // N * 32 = 2M threads
  int n = id >> 5, j = id & 31;
  int kidx = (int)out[OUT_IDX + n];
  kidx = (kidx < 0) ? 0 : (kidx >= KCODES ? KCODES - 1 : kidx);
  int t = n >> 4, m = n & 15, c = j >> 2, g = j & 3;
  int lane = m + (g >> 1) * 16;
  int e0 = (g & 1) ? 8 : 0;
  size_t elem = ((size_t)(t * 8 + c) * 32 + lane) * 16 + e0;
  uint4 u4 = *(const uint4*)(wsb + OFF_XN + elem * 2);
  unsigned o[4] = {u4.x, u4.y, u4.z, u4.w};
  float* dst = out + OUT_EMAW + (size_t)kidx * DFEAT + j * 8;
#pragma unroll
  for (int i = 0; i < 4; ++i) {
    atomAddF(&dst[2 * i],     OMD * bf16lo(o[i]));
    atomAddF(&dst[2 * i + 1], OMD * bf16hi(o[i]));
  }
  if (j == 0) atomAddF(&ws[WS_CNT + kidx], 1.0f);
}

// 9) cs_pre = decay*cs + (1-decay)*counts ; n_total = sum(cs_pre)
__global__ void k_cs_pre(const float* __restrict__ cs, float* ws) {
  int k = blockIdx.x * 256 + threadIdx.x;
  float v = DECAY * cs[k] + OMD * ws[WS_CNT + k];
  ws[WS_CSPRE + k] = v;
  __shared__ float sm[256];
  sm[threadIdx.x] = v;
  __syncthreads();
  for (int s = 128; s > 0; s >>= 1) {
    if (threadIdx.x < s) sm[threadIdx.x] += sm[threadIdx.x + s];
    __syncthreads();
  }
  if (threadIdx.x == 0) atomAddF(&ws[WS_NTOT], sm[0]);
}

// 10) final elementwise tail: new_cs, new_embedding, embedding_output
__global__ void k_final(const float* ws, float* out) {
  int id = blockIdx.x * 256 + threadIdx.x;    // K*D = 1M threads
  int k = id >> 8, d = id & 255;
  float nt = ws[WS_NTOT];
  float csf = (ws[WS_CSPRE + k] + 1e-5f) / (nt + (float)KCODES * 1e-5f) * nt;
  float w = out[OUT_EMAW + id];
  float emb = w / csf;
  out[OUT_EMB + id]  = emb;
  out[OUT_EMBO + id] = emb * ws[WS_RSTD + d] + out[OUT_RM + d];
  if (d == 0) out[OUT_CS + k] = csf;
}

// ---------------------------------------------------------------------------
extern "C" void kernel_launch(void* const* d_in, const int* in_sizes, int n_in,
                              void* d_out, int out_size, void* d_ws, size_t ws_size,
                              hipStream_t stream) {
  const float* x     = (const float*)d_in[0];
  const float* emb   = (const float*)d_in[1];
  const float* emacs = (const float*)d_in[2];
  const float* emaw  = (const float*)d_in[3];
  const float* rmean = (const float*)d_in[4];
  const float* rvar  = (const float*)d_in[5];
  float* out = (float*)d_out;
  float* ws = (float*)d_ws;
  unsigned char* wsb = (unsigned char*)d_ws;
  (void)in_sizes; (void)n_in; (void)out_size; (void)ws_size;

  k_init_ws<<<(WS_HEADF + 255) / 256, 256, 0, stream>>>(ws);
  k_bn_stats<<<N_ROWS / 256, 256, 0, stream>>>(x, ws);
  k_bn_finalize<<<1, 256, 0, stream>>>(ws, rmean, rvar, out);
  k_swizzle_x<<<(N_ROWS * (DFEAT / 8)) / 256, 256, 0, stream>>>(x, ws, wsb);
  k_swizzle_e<<<(KCODES * (DFEAT / 16)) / 256, 256, 0, stream>>>(emb, ws, wsb);
  k_ema_init<<<(KCODES * DFEAT) / 256, 256, 0, stream>>>(emaw, out);
  k_gemm_argmin<<<N_ROWS / 128, dim3(32, 8), 0, stream>>>(wsb, ws, out);
  k_scatter<<<(N_ROWS * 32) / 256, 256, 0, stream>>>(wsb, ws, out);
  k_cs_pre<<<KCODES / 256, 256, 0, stream>>>(emacs, ws);
  k_final<<<(KCODES * DFEAT) / 256, 256, 0, stream>>>(ws, out);
}
